// NTM_Head_39281770889800
// MI455X (gfx1250) — matrix operand order repaired and checked by reading, not compiled, not run on hardware
//
#include <hip/hip_runtime.h>

// ---------------------------------------------------------------------------
// NTM head forward, MI455X (gfx1250, wave32).
// Roofline: bank = 256MB fp32. Fused rank-1 new_bank update => 3 streaming
// passes (~768MB @ 23.3TB/s ~= 33us floor). Controller GEMMs (~100 MFLOP) on
// v_wmma_f32_16x16x32_f16 with split hi/lo f16 (3 WMMAs) for ~fp32 accuracy.
// Deterministic: no atomics; all reductions fixed-order partials.
// ---------------------------------------------------------------------------

#define BB    128
#define NN    8192
#define MM    64
#define HIDD  512
#define OUTC  256
#define REPR_LD 80   // 70 valid cols padded to 80
#define SEQW  63

typedef __attribute__((ext_vector_type(16))) _Float16 v16h;
typedef __attribute__((ext_vector_type(8)))  float    v8f;

__device__ __forceinline__ float wred_sum(float v) {
#pragma unroll
  for (int off = 16; off; off >>= 1) v += __shfl_xor(v, off, 32);
  return v;
}
__device__ __forceinline__ float wred_max(float v) {
#pragma unroll
  for (int off = 16; off; off >>= 1) v = fmaxf(v, __shfl_xor(v, off, 32));
  return v;
}

template <bool IS_MAX>
__device__ __forceinline__ float block_red(float v, float* sm) {
  const int lane = threadIdx.x & 31, wid = threadIdx.x >> 5;
  const int nw = blockDim.x >> 5;
  v = IS_MAX ? wred_max(v) : wred_sum(v);
  if (lane == 0) sm[wid] = v;
  __syncthreads();
  if (wid == 0) {
    float r = (lane < nw) ? sm[lane] : (IS_MAX ? -3.402823e38f : 0.f);
    r = IS_MAX ? wred_max(r) : wred_sum(r);
    if (lane == 0) sm[0] = r;
  }
  __syncthreads();
  float out = sm[0];
  __syncthreads();
  return out;
}

__device__ __forceinline__ float sigmoidf(float x) { return 1.f / (1.f + expf(-x)); }

// ---------------------------------------------------------------------------
// C = act(A[rows x K] @ W[K x Ncols] + bias). One wave per 16x16 tile.
// Split-precision f16 WMMA (hi/lo), fp32 accumulate. K % 32 == 0, rows % 16 == 0.
// ACT: 0=id, 1=leaky_relu(0.01), 2=sigmoid.
// ---------------------------------------------------------------------------
template <int ACT>
__global__ void __launch_bounds__(32)
gemm_wmma_kernel(const float* __restrict__ A, int ldA,
                 const float* __restrict__ W, int ldW,
                 const float* __restrict__ bias,
                 float* __restrict__ C, int ldC,
                 int Ncols, int K) {
  const int lane = threadIdx.x;
  const int half = lane >> 4, l16 = lane & 15;
  const int rowA = blockIdx.y * 16 + l16;
  const int colB = blockIdx.x * 16 + l16;
  v8f acc = {};
  for (int k0 = 0; k0 < K; k0 += 32) {
    v16h ah, al, bh, bl;
    // A (16x32 f16 layout): lanes0-15 K=[0..7],[16..23]; lanes16-31 +8
    const float* Ap = A + (size_t)rowA * ldA + k0 + 8 * half;
#pragma unroll
    for (int i = 0; i < 8; ++i) {
      const float a0 = Ap[i];
      const _Float16 h0 = (_Float16)a0;
      ah[i] = h0;
      al[i] = (_Float16)(a0 - (float)h0);
      const float a1 = Ap[16 + i];
      const _Float16 h1 = (_Float16)a1;
      ah[8 + i] = h1;
      al[8 + i] = (_Float16)(a1 - (float)h1);
    }
    // B (32x16): lanes0-15 K=[0..7],[16..23]; lanes16-31 +8 (matches A)
#pragma unroll
    for (int i = 0; i < 16; ++i) {
      const int kk = k0 + 8 * half + (i & 7) + 16 * (i >> 3);
      const float w = (colB < Ncols) ? W[(size_t)kk * ldW + colB] : 0.f;
      const _Float16 hw = (_Float16)w;
      bh[i] = hw;
      bl[i] = (_Float16)(w - (float)hw);
    }
    acc = __builtin_amdgcn_wmma_f32_16x16x32_f16(false, ah, false, bh, (short)0, acc, false, false);
    acc = __builtin_amdgcn_wmma_f32_16x16x32_f16(false, ah, false, bl, (short)0, acc, false, false);
    acc = __builtin_amdgcn_wmma_f32_16x16x32_f16(false, al, false, bh, (short)0, acc, false, false);
  }
  if (colB < Ncols) {
    const float bv = bias[colB];
#pragma unroll
    for (int v = 0; v < 8; ++v) {
      const int mr = blockIdx.y * 16 + v + 8 * half;  // C/D layout: m = v + 8*half
      float val = acc[v] + bv;
      if (ACT == 1) val = (val > 0.f) ? val : 0.01f * val;
      if (ACT == 2) val = sigmoidf(val);
      C[(size_t)mr * ldC + colB] = val;
    }
  }
}

// ---------------------------------------------------------------------------
// Per-batch sums of the (unnormalized) previous read/write weights.
// ---------------------------------------------------------------------------
__global__ void __launch_bounds__(256)
reduce_rows(const float* __restrict__ a, const float* __restrict__ b2,
            float* __restrict__ outa, float* __restrict__ outb) {
  __shared__ float sm[32];
  const int bb = blockIdx.x;
  float sa = 0.f, sb = 0.f;
  for (int n = threadIdx.x; n < NN; n += 256) {
    sa += a[(size_t)bb * NN + n];
    sb += b2[(size_t)bb * NN + n];
  }
  sa = block_red<false>(sa, sm);
  sb = block_red<false>(sb, sm);
  if (threadIdx.x == 0) { outa[bb] = sa; outb[bb] = sb; }
}

// ---------------------------------------------------------------------------
// ||kr||, ||kw|| per batch (k = repr cols 0..63).
// ---------------------------------------------------------------------------
__global__ void __launch_bounds__(32)
knorm_kernel(const float* __restrict__ reprR, const float* __restrict__ reprW,
             float* __restrict__ knR, float* __restrict__ knW) {
  const int b = blockIdx.x, lane = threadIdx.x;
  float r0 = reprR[(size_t)b * REPR_LD + lane], r1 = reprR[(size_t)b * REPR_LD + lane + 32];
  float w0 = reprW[(size_t)b * REPR_LD + lane], w1 = reprW[(size_t)b * REPR_LD + lane + 32];
  float sr = wred_sum(r0 * r0 + r1 * r1);
  float sw = wred_sum(w0 * w0 + w1 * w1);
  if (lane == 0) { knR[b] = sqrtf(sr); knW[b] = sqrtf(sw); }
}

// ---------------------------------------------------------------------------
// Bank pass 1: per row n -> dot(row,kw), dot(row,kr), ||row||^2; and fixed-order
// per-block partial column sums for mem_avg. Wave-per-row, lane = column.
// ---------------------------------------------------------------------------
__global__ void __launch_bounds__(256)
bank_pass1(const float* __restrict__ bank,
           const float* __restrict__ reprW, const float* __restrict__ reprR,
           float* __restrict__ dotW, float* __restrict__ dotR,
           float* __restrict__ sqn, float* __restrict__ memsum_part) {
  __shared__ float red[8][64];
  const int b = blockIdx.y;
  const int lane = threadIdx.x & 31, wave = threadIdx.x >> 5;
  const int n0 = blockIdx.x * 256 + wave * 32;
  const float kw0 = reprW[(size_t)b * REPR_LD + lane];
  const float kw1 = reprW[(size_t)b * REPR_LD + lane + 32];
  const float kr0 = reprR[(size_t)b * REPR_LD + lane];
  const float kr1 = reprR[(size_t)b * REPR_LD + lane + 32];
  const float* base = bank + (size_t)b * NN * MM;
  float ms0 = 0.f, ms1 = 0.f;
  for (int i = 0; i < 32; ++i) {
    const int n = n0 + i;
    const float* row = base + (size_t)n * MM;
    const float x0 = row[lane], x1 = row[lane + 32];
    ms0 += x0; ms1 += x1;
    float pw = x0 * kw0 + x1 * kw1;
    float pr = x0 * kr0 + x1 * kr1;
    float ps = x0 * x0 + x1 * x1;
    pw = wred_sum(pw); pr = wred_sum(pr); ps = wred_sum(ps);
    if (lane == 0) {
      dotW[(size_t)b * NN + n] = pw;
      dotR[(size_t)b * NN + n] = pr;
      sqn [(size_t)b * NN + n] = ps;
    }
  }
  red[wave][lane] = ms0;
  red[wave][lane + 32] = ms1;
  __syncthreads();
  if (threadIdx.x < 64) {
    float s = 0.f;
#pragma unroll
    for (int w = 0; w < 8; ++w) s += red[w][threadIdx.x];
    memsum_part[((size_t)b * 32 + blockIdx.x) * 64 + threadIdx.x] = s;
  }
}

// ---------------------------------------------------------------------------
// Build concat(mem_avg, kw) for the erase/add GEMM (fixed-order partial sum).
// ---------------------------------------------------------------------------
__global__ void __launch_bounds__(128)
ea_prep(const float* __restrict__ memsum_part, const float* __restrict__ reprW,
        float* __restrict__ concatb) {
  const int b = blockIdx.x, j = threadIdx.x;  // 128 threads
  float v;
  if (j < 64) {
    float s = 0.f;
    for (int p = 0; p < 32; ++p) s += memsum_part[((size_t)b * 32 + p) * 64 + j];
    v = s * (1.f / (float)NN);
  } else {
    v = reprW[(size_t)b * REPR_LD + (j - 64)];
  }
  concatb[b * 128 + j] = v;
}

// ---------------------------------------------------------------------------
// ae_diff = a - e = ea[64:] - sigmoid(ea[:64]); plus ||ae||^2 and dot(ae, kr).
// ---------------------------------------------------------------------------
__global__ void __launch_bounds__(32)
ae_finalize(const float* __restrict__ ea, const float* __restrict__ reprR,
            float* __restrict__ ae_diff, float* __restrict__ ae_nsq,
            float* __restrict__ ae_dotkr) {
  const int b = blockIdx.x, lane = threadIdx.x;
  const float d0 = ea[b * 128 + 64 + lane]      - sigmoidf(ea[b * 128 + lane]);
  const float d1 = ea[b * 128 + 64 + lane + 32] - sigmoidf(ea[b * 128 + lane + 32]);
  ae_diff[b * 64 + lane] = d0;
  ae_diff[b * 64 + lane + 32] = d1;
  float ns = wred_sum(d0 * d0 + d1 * d1);
  float dk = wred_sum(d0 * reprR[(size_t)b * REPR_LD + lane] +
                      d1 * reprR[(size_t)b * REPR_LD + lane + 32]);
  if (lane == 0) { ae_nsq[b] = ns; ae_dotkr[b] = dk; }
}

// ---------------------------------------------------------------------------
// NTM addressing: cos -> softmax(beta*cos) -> interpolate -> shift -> sharpen.
// One block per batch; logits in registers; wg staged in LDS for the +/-1
// circular shift. ADJ folds the rank-1 new_bank correction for the read head.
// ---------------------------------------------------------------------------
template <bool ADJ>
__global__ void __launch_bounds__(1024)
address_kernel(const float* __restrict__ dotk,  const float* __restrict__ sqn,
               const float* __restrict__ w_prev, const float* __restrict__ wprev_sum,
               const float* __restrict__ rep,    const float* __restrict__ knorm,
               const float* __restrict__ w_w,    const float* __restrict__ dotae,
               const float* __restrict__ ae_dot_k, const float* __restrict__ ae_nsq,
               float* __restrict__ w_out) {
  __shared__ float sWg[NN];
  __shared__ float sred[32];
  const int b = blockIdx.x;
  const int tid = threadIdx.x;
  const float* r = rep + (size_t)b * REPR_LD;
  const float beta  = fmaxf(r[64], 0.f) + 1e-8f;
  const float g     = sigmoidf(r[65]);
  const float s0r = r[66], s1r = r[67], s2r = r[68];
  const float gamma = fmaxf(r[69], 0.f) + 1.f;
  const float smx = fmaxf(fmaxf(s0r, s1r), s2r);
  const float e0 = expf(s0r - smx), e1 = expf(s1r - smx), e2 = expf(s2r - smx);
  const float es = e0 + e1 + e2;
  const float s0 = e0 / es, s1 = e1 / es, s2 = e2 / es;
  const float kn = knorm[b];
  const float invwp = 1.f / wprev_sum[b];
  const float adk  = ADJ ? ae_dot_k[b] : 0.f;
  const float ansq = ADJ ? ae_nsq[b]   : 0.f;
  const size_t base = (size_t)b * NN;

  float logit[NN / 1024];
  float lmax = -3.402823e38f;
#pragma unroll
  for (int i = 0; i < NN / 1024; ++i) {
    const int n = tid + i * 1024;
    float d = dotk[base + n];
    float q = sqn[base + n];
    if (ADJ) {
      const float ww = w_w[base + n];
      d += ww * adk;
      q += 2.f * ww * dotae[base + n] + ww * ww * ansq;
    }
    const float cs = d / fmaxf(sqrtf(q) * kn, 1e-8f);
    logit[i] = beta * cs;
    lmax = fmaxf(lmax, logit[i]);
  }
  const float bmax = block_red<true>(lmax, sred);
  float lsum = 0.f;
#pragma unroll
  for (int i = 0; i < NN / 1024; ++i) { logit[i] = expf(logit[i] - bmax); lsum += logit[i]; }
  const float bsum = block_red<false>(lsum, sred);
  const float inv = 1.f / bsum;
#pragma unroll
  for (int i = 0; i < NN / 1024; ++i) {
    const int n = tid + i * 1024;
    const float wc = logit[i] * inv;
    sWg[n] = g * wc + (1.f - g) * (w_prev[base + n] * invwp);
  }
  __syncthreads();
  float psum = 0.f;
  float wp[NN / 1024];
#pragma unroll
  for (int i = 0; i < NN / 1024; ++i) {
    const int n = tid + i * 1024;
    const int nm = (n == 0) ? NN - 1 : n - 1;
    const int np = (n == NN - 1) ? 0 : n + 1;
    const float wsv = s0 * sWg[nm] + s1 * sWg[n] + s2 * sWg[np];
    wp[i] = powf(wsv, gamma);
    psum += wp[i];
  }
  const float bp = block_red<false>(psum, sred);
  const float invp = 1.f / bp;
#pragma unroll
  for (int i = 0; i < NN / 1024; ++i) {
    const int n = tid + i * 1024;
    w_out[base + n] = wp[i] * invp;
  }
}

// ---------------------------------------------------------------------------
// Bank pass 2: dot(row, a-e) per row.
// ---------------------------------------------------------------------------
__global__ void __launch_bounds__(256)
bank_pass2(const float* __restrict__ bank, const float* __restrict__ ae,
           float* __restrict__ dotae) {
  const int b = blockIdx.y;
  const int lane = threadIdx.x & 31, wave = threadIdx.x >> 5;
  const int n0 = blockIdx.x * 256 + wave * 32;
  const float a0 = ae[b * 64 + lane], a1 = ae[b * 64 + lane + 32];
  const float* base = bank + (size_t)b * NN * MM;
  for (int i = 0; i < 32; ++i) {
    const int n = n0 + i;
    const float* row = base + (size_t)n * MM;
    float p = row[lane] * a0 + row[lane + 32] * a1;
    p = wred_sum(p);
    if (lane == 0) dotae[(size_t)b * NN + n] = p;
  }
}

// ---------------------------------------------------------------------------
// Bank pass 3: out_partial = sum_n w_r[n]*row; wrww_partial = sum_n w_r*w_w.
// ---------------------------------------------------------------------------
__global__ void __launch_bounds__(256)
bank_pass3(const float* __restrict__ bank, const float* __restrict__ w_r,
           const float* __restrict__ w_w, float* __restrict__ out_part,
           float* __restrict__ wrww_part) {
  __shared__ float red[8][64];
  __shared__ float red2[8];
  const int b = blockIdx.y;
  const int lane = threadIdx.x & 31, wave = threadIdx.x >> 5;
  const int n0 = blockIdx.x * 256 + wave * 32;
  const float* base = bank + (size_t)b * NN * MM;
  float o0 = 0.f, o1 = 0.f, sw = 0.f;
  for (int i = 0; i < 32; ++i) {
    const int n = n0 + i;
    const float* row = base + (size_t)n * MM;
    const float wr = w_r[(size_t)b * NN + n];
    o0 += wr * row[lane];
    o1 += wr * row[lane + 32];
    sw += wr * w_w[(size_t)b * NN + n];
  }
  red[wave][lane] = o0;
  red[wave][lane + 32] = o1;
  if (lane == 0) red2[wave] = sw;
  __syncthreads();
  if (threadIdx.x < 64) {
    float s = 0.f;
#pragma unroll
    for (int w = 0; w < 8; ++w) s += red[w][threadIdx.x];
    out_part[((size_t)b * 32 + blockIdx.x) * 64 + threadIdx.x] = s;
  }
  if (threadIdx.x == 0) {
    float s = 0.f;
#pragma unroll
    for (int w = 0; w < 8; ++w) s += red2[w];
    wrww_part[b * 32 + blockIdx.x] = s;
  }
}

// ---------------------------------------------------------------------------
// out[b][m] = sum_p out_part + (sum_p wrww_part) * ae_diff[b][m]
// ---------------------------------------------------------------------------
__global__ void __launch_bounds__(64)
out_finalize(const float* __restrict__ out_part, const float* __restrict__ wrww_part,
             const float* __restrict__ ae_diff, float* __restrict__ out_vec) {
  const int b = blockIdx.x, m = threadIdx.x;  // 64
  float s = 0.f, w = 0.f;
  for (int p = 0; p < 32; ++p) {
    s += out_part[((size_t)b * 32 + p) * 64 + m];
    w += wrww_part[b * 32 + p];
  }
  out_vec[b * 64 + m] = s + w * ae_diff[b * 64 + m];
}

// ---------------------------------------------------------------------------

extern "C" void kernel_launch(void* const* d_in, const int* in_sizes, int n_in,
                              void* d_out, int out_size, void* d_ws, size_t ws_size,
                              hipStream_t stream) {
  const float* x    = (const float*)d_in[0];
  const float* bank = (const float*)d_in[1];
  const float* wrp  = (const float*)d_in[2];
  const float* wwp  = (const float*)d_in[3];
  const float* W0   = (const float*)d_in[4];
  const float* b0   = (const float*)d_in[5];
  const float* W1   = (const float*)d_in[6];
  const float* b1   = (const float*)d_in[7];
  const float* Wc   = (const float*)d_in[8];
  const float* bc   = (const float*)d_in[9];
  const float* Wr   = (const float*)d_in[10];
  const float* br   = (const float*)d_in[11];
  const float* Ww   = (const float*)d_in[12];
  const float* bw   = (const float*)d_in[13];
  const float* Wea  = (const float*)d_in[14];
  const float* bea  = (const float*)d_in[15];
  const float* Wsp  = (const float*)d_in[16];
  const float* bsp  = (const float*)d_in[17];
  const float* Wo   = (const float*)d_in[18];
  const float* bo   = (const float*)d_in[19];
  (void)in_sizes; (void)n_in; (void)out_size; (void)ws_size;

  float* ws = (float*)d_ws;
  size_t off = 0;
  auto take = [&](size_t n) { float* p = ws + off; off += n; return p; };
  float* h0      = take((size_t)BB * HIDD);
  float* h1      = take((size_t)BB * HIDD);
  float* ctrl    = take((size_t)BB * OUTC);
  float* reprR   = take((size_t)BB * REPR_LD);
  float* reprW   = take((size_t)BB * REPR_LD);
  float* catb    = take((size_t)BB * 128);
  float* eab     = take((size_t)BB * 128);
  float* aed     = take((size_t)BB * 64);
  float* outv    = take((size_t)BB * 64);
  float* seqb    = take((size_t)BB * HIDD);
  float* wsumR   = take(BB);
  float* wsumW   = take(BB);
  float* knR     = take(BB);
  float* knW     = take(BB);
  float* aeNsq   = take(BB);
  float* aeDotK  = take(BB);
  float* msPart  = take((size_t)BB * 32 * 64);
  float* outPart = take((size_t)BB * 32 * 64);
  float* wrwwP   = take((size_t)BB * 32);
  float* dotW    = take((size_t)BB * NN);
  float* dotR    = take((size_t)BB * NN);
  float* sqn     = take((size_t)BB * NN);
  float* dotAE   = take((size_t)BB * NN);
  float* w_w     = take((size_t)BB * NN);
  float* w_r     = take((size_t)BB * NN);

  // previous-weight normalizers
  reduce_rows<<<BB, 256, 0, stream>>>(wrp, wwp, wsumR, wsumW);

  // controller MLP (WMMA)
  {
    dim3 g(HIDD / 16, BB / 16);
    gemm_wmma_kernel<1><<<g, 32, 0, stream>>>(x, 64, W0, HIDD, b0, h0, HIDD, HIDD, 64);
    gemm_wmma_kernel<1><<<g, 32, 0, stream>>>(h0, HIDD, W1, HIDD, b1, h1, HIDD, HIDD, HIDD);
  }
  {
    dim3 g(OUTC / 16, BB / 16);
    gemm_wmma_kernel<1><<<g, 32, 0, stream>>>(h1, HIDD, Wc, OUTC, bc, ctrl, OUTC, OUTC, HIDD);
  }
  {
    dim3 g(5, BB / 16);  // 70 cols -> 5 tiles of 16, store-guarded
    gemm_wmma_kernel<0><<<g, 32, 0, stream>>>(ctrl, OUTC, Wr, 70, br, reprR, REPR_LD, 70, OUTC);
    gemm_wmma_kernel<0><<<g, 32, 0, stream>>>(ctrl, OUTC, Ww, 70, bw, reprW, REPR_LD, 70, OUTC);
  }
  knorm_kernel<<<BB, 32, 0, stream>>>(reprR, reprW, knR, knW);

  // bank pass 1 (dot_kw, dot_kr, ||row||^2, mem-sum partials)
  {
    dim3 g(NN / 256, BB);
    bank_pass1<<<g, 256, 0, stream>>>(bank, reprW, reprR, dotW, dotR, sqn, msPart);
  }
  ea_prep<<<BB, 128, 0, stream>>>(msPart, reprW, catb);
  {
    dim3 g(128 / 16, BB / 16);
    gemm_wmma_kernel<0><<<g, 32, 0, stream>>>(catb, 128, Wea, 128, bea, eab, 128, 128, 128);
  }
  ae_finalize<<<BB, 32, 0, stream>>>(eab, reprR, aed, aeNsq, aeDotK);

  // write-head addressing
  address_kernel<false><<<BB, 1024, 0, stream>>>(dotW, sqn, wwp, wsumW, reprW, knW,
                                                 nullptr, nullptr, nullptr, nullptr, w_w);
  // bank pass 2 (dot(row, a-e))
  {
    dim3 g(NN / 256, BB);
    bank_pass2<<<g, 256, 0, stream>>>(bank, aed, dotAE);
  }
  // read-head addressing on implicit new_bank (rank-1 folded)
  address_kernel<true><<<BB, 1024, 0, stream>>>(dotR, sqn, wrp, wsumR, reprR, knR,
                                                w_w, dotAE, aeDotK, aeNsq, w_r);
  // bank pass 3 (weighted read-out)
  {
    dim3 g(NN / 256, BB);
    bank_pass3<<<g, 256, 0, stream>>>(bank, w_r, w_w, outPart, wrwwP);
  }
  out_finalize<<<BB, 64, 0, stream>>>(outPart, wrwwP, aed, outv);

  // output MLP (WMMA)
  {
    dim3 g(HIDD / 16, BB / 16);
    gemm_wmma_kernel<1><<<g, 32, 0, stream>>>(outv, 64, Wsp, HIDD, bsp, seqb, HIDD, HIDD, 64);
  }
  {
    dim3 g(4, BB / 16);  // 63 cols -> 4 tiles, store-guarded
    gemm_wmma_kernel<2><<<g, 32, 0, stream>>>(seqb, HIDD, Wo, SEQW, bo, (float*)d_out, SEQW, SEQW, HIDD);
  }
}
